// EAGFM_74929999446431
// MI455X (gfx1250) — matrix operand order repaired and checked by reading, not compiled, not run on hardware
//
#include <hip/hip_runtime.h>

// ---------------------------------------------------------------------------
// EAGFM fused implementation for gfx1250 (MI455X).
// All GEMM work goes through v_wmma_f32_16x16x32_bf16 (wave32 WMMA).
// Attention Q/K staging uses GLOBAL_LOAD_ASYNC_TO_LDS_B128 (ASYNCcnt path).
// bf16 is stored as raw unsigned short (bit pattern); conversions are manual.
// ---------------------------------------------------------------------------

typedef __attribute__((ext_vector_type(16))) __bf16 v16bf;
typedef __attribute__((ext_vector_type(8)))  float  v8f;

#define Bn   16
#define Cc   256
#define Hh   80
#define Ww   80
#define HWp  6400
#define SEQ  80
#define KP   96      // SEQ padded to multiple of 32 for PV GEMM K dim

__device__ __forceinline__ unsigned short f2bf(float f) {
    unsigned int u = __float_as_uint(f);
    // round-to-nearest-even
    unsigned int r = (u + 0x7FFFu + ((u >> 16) & 1u)) >> 16;
    return (unsigned short)r;
}
__device__ __forceinline__ float bf2f(unsigned short h) {
    return __uint_as_float(((unsigned int)h) << 16);
}

union Frag16 { v16bf v; unsigned int u[8]; };

__device__ __forceinline__ v8f vzero() {
    v8f z = {0.f, 0.f, 0.f, 0.f, 0.f, 0.f, 0.f, 0.f};
    return z;
}

// Async bulk copy: 16 bytes per lane, global -> LDS, tracked by ASYNCcnt.
// VDST VGPR carries the per-lane LDS byte offset (ISA 08_async_tensor §4.4,
// GV mode: the 16 bytes at global VADDR go to LDS at that offset).
__device__ __forceinline__ void async_copy16(unsigned lds_byte_off, const void* gaddr) {
    asm volatile("global_load_async_to_lds_b128 %0, %1, off"
                 :: "v"(lds_byte_off), "v"(gaddr)
                 : "memory");
}
__device__ __forceinline__ void wait_async0() {
    asm volatile("s_wait_asynccnt 0x0" ::: "memory");
}

// A fragment: 16x32 bf16, row-major tile base (row stride lda elements).
// ISA 7.12.2: lane m=L%16, khalf=L/16; VGPR v: k0=(v/4)*16+khalf*8+(v%4)*2.
__device__ __forceinline__ v16bf load_frag_a(const unsigned short* base, int lda) {
    const int lane = threadIdx.x & 31;
    const int m = lane & 15, kh = lane >> 4;
    Frag16 f;
#pragma unroll
    for (int v = 0; v < 8; ++v) {
        int k0 = (v >> 2) * 16 + kh * 8 + (v & 3) * 2;
        f.u[v] = *(const unsigned int*)(base + m * lda + k0);
    }
    return f.v;
}

// B fragment: 32x16 bf16, staged n-major (Bs[n][k], row stride ldb).
// lane n=L%16, khalf=L/16; VGPR v: k0=(v/4)*16+khalf*8+(v%4)*2.
__device__ __forceinline__ v16bf load_frag_b(const unsigned short* base, int ldb) {
    const int lane = threadIdx.x & 31;
    const int n = lane & 15, kh = lane >> 4;
    Frag16 f;
#pragma unroll
    for (int v = 0; v < 8; ++v) {
        int k0 = (v >> 2) * 16 + kh * 8 + (v & 3) * 2;
        f.u[v] = *(const unsigned int*)(base + n * ldb + k0);
    }
    return f.v;
}

__device__ __forceinline__ v8f wmma_bf16(v16bf a, v16bf b, v8f c) {
    return __builtin_amdgcn_wmma_f32_16x16x32_bf16(
        /*neg_a=*/false, a, /*neg_b=*/false, b,
        /*c_mod=*/(short)0, c, /*reuse_a=*/false, /*reuse_b=*/false);
}

// ---------------------------------------------------------------------------
// Kernel 1: conv1x1 as GEMM.  Y[b, co, p] = sum_ci W[co,ci] * X[b,ci,p] + b[co]
// Block tile 128(M) x 64(N), K-step 32, 8 waves (wave = 32x32 sub-tile).
// ---------------------------------------------------------------------------
__global__ __launch_bounds__(256) void conv1x1_kernel(
    const float* __restrict__ X,      // [B, C, HW] f32
    const float* __restrict__ Wm,     // [C, C] f32
    const float* __restrict__ bias,   // [C]
    unsigned short* __restrict__ Y)   // [B, C, HW] bf16
{
    __shared__ unsigned short As[128 * 32];
    __shared__ unsigned short Bs[64 * 32];   // Bs[n][k]

    const int t   = threadIdx.x;
    const int wv  = t >> 5;
    const int n0  = blockIdx.x * 64;
    const int m0  = blockIdx.y * 128;
    const int b   = blockIdx.z;
    const float* Xb = X + (size_t)b * Cc * HWp;

    const int wm = wv >> 1;   // 0..3 -> 32 rows each
    const int wn = wv & 1;    // 0..1 -> 32 cols each

    v8f acc[2][2];
    acc[0][0] = vzero(); acc[0][1] = vzero();
    acc[1][0] = vzero(); acc[1][1] = vzero();

    for (int kb = 0; kb < Cc; kb += 32) {
        // Stage A (weights), k-contiguous coalesced reads.
#pragma unroll
        for (int i = 0; i < 16; ++i) {
            int e = t + i * 256;            // 0..4095
            int m = e >> 5, k = e & 31;
            As[e] = f2bf(Wm[(m0 + m) * Cc + kb + k]);
        }
        // Stage B n-major: Bs[n][k] = X[kb+k][n0+n]; n-fastest coalesced reads.
#pragma unroll
        for (int i = 0; i < 8; ++i) {
            int e = t + i * 256;            // 0..2047
            int k = e >> 6, n = e & 63;
            Bs[n * 32 + k] = f2bf(Xb[(size_t)(kb + k) * HWp + n0 + n]);
        }
        if (kb + 32 < Cc) {                  // gfx1250 global_prefetch_b8
            __builtin_prefetch(&Wm[(m0 + (t >> 1)) * Cc + kb + 32], 0, 1);
            __builtin_prefetch(&Xb[(size_t)(kb + 32 + (t & 31)) * HWp + n0], 0, 1);
        }
        __syncthreads();
#pragma unroll
        for (int i = 0; i < 2; ++i) {
            v16bf af = load_frag_a(As + (wm * 32 + i * 16) * 32, 32);
#pragma unroll
            for (int j = 0; j < 2; ++j) {
                v16bf bf = load_frag_b(Bs + (wn * 32 + j * 16) * 32, 32);
                acc[i][j] = wmma_bf16(af, bf, acc[i][j]);
            }
        }
        __syncthreads();
    }

    const int lane = t & 31;
    const int ml = (lane >> 4) * 8;
    const int nl = lane & 15;
#pragma unroll
    for (int i = 0; i < 2; ++i) {
#pragma unroll
        for (int j = 0; j < 2; ++j) {
            const int mb2 = m0 + wm * 32 + i * 16;
            const int nb2 = n0 + wn * 32 + j * 16;
#pragma unroll
            for (int r = 0; r < 8; ++r) {
                int m = mb2 + ml + r;
                int n = nb2 + nl;
                float val = acc[i][j][r] + bias[m];
                Y[(size_t)b * Cc * HWp + (size_t)m * HWp + n] = f2bf(val);
            }
        }
    }
}

// ---------------------------------------------------------------------------
// Kernel 2: depthwise 3x3 (+bias), NCHW bf16 in -> [B,H,W,C] bf16 out.
// Block covers 16 channels x 16 w at one h; LDS-staged input with halo.
// ---------------------------------------------------------------------------
__global__ __launch_bounds__(256) void dw3x3_kernel(
    const unsigned short* __restrict__ Yin,  // [B,C,H,W] bf16
    const float* __restrict__ w2,            // [C,1,3,3]
    const float* __restrict__ b2,            // [C]
    unsigned short* __restrict__ Out)        // [B,H,W,C] bf16
{
    __shared__ unsigned short sIn[16][3][18];
    const int t  = threadIdx.x;
    const int w0 = blockIdx.x * 16;
    const int h  = blockIdx.y;
    const int b  = blockIdx.z >> 4;
    const int c0 = (blockIdx.z & 15) * 16;

    for (int e = t; e < 16 * 3 * 18; e += 256) {
        int ch = e / 54, rem = e % 54;
        int dy = rem / 18, dx = rem % 18;
        int y = h + dy - 1, x = w0 + dx - 1;
        unsigned short v = 0;
        if (y >= 0 && y < Hh && x >= 0 && x < Ww)
            v = Yin[((size_t)(b * Cc + c0 + ch)) * HWp + y * Ww + x];
        sIn[ch][dy][dx] = v;
    }
    __syncthreads();

    const int c  = t & 15;   // c fastest -> coalesced [.,c] writes
    const int wl = t >> 4;
    const float* wp = w2 + (c0 + c) * 9;
    float s = b2[c0 + c];
#pragma unroll
    for (int dy = 0; dy < 3; ++dy)
#pragma unroll
        for (int dx = 0; dx < 3; ++dx)
            s += wp[dy * 3 + dx] * bf2f(sIn[c][dy][wl + dx]);
    Out[(((size_t)(b * Hh + h)) * Ww + (w0 + wl)) * Cc + c0 + c] = f2bf(s);
}

// ---------------------------------------------------------------------------
// Kernel 3: per-(b,h) row attention, fully fused through the lp3/rp3 output
// projections.  One workgroup (8 waves) per (b,h); everything lives in LDS.
//   attn = Ql Kr^T * 1/16            (WMMA, 5x5 tiles, K=256)
//   Pr = row-softmax(attn), Pc = col-softmax(attn) stored transposed
//   Fr2l = Pr @ Vr, Fl2r = Pc @ Vl   (WMMA, K padded 80->96)
//   pattn1[:,w] = lp3 @ Fr2l^T + rp3 @ Fl2r^T + biases  (WMMA, K=256)
// LDS map (dynamic, 236544 B; no static __shared__ in this kernel, so the
// dynamic region starts at LDS byte offset 0 -> the offsets below are the
// hardware dsaddr values used by the async loads):
//   sQ   [80][256] bf16  @0       -> reused as Fr2l
//   sK   [80][256] bf16  @40960   -> reused as Fl2r
//   sVrT [256][96] bf16  @81920
//   sVlT [256][96] bf16  @131072
//   attn [80][80]  f32   @180224  -> reused as weight tile As[256][32] bf16
//   sPr  [80][96]  bf16  @205824
//   sPc  [80][96]  bf16  @221184
// ---------------------------------------------------------------------------
#define ATTN_LDS_BYTES 236544
#define SK_LDS_OFF 40960u

__global__ __launch_bounds__(256) void attn_kernel(
    const unsigned short* __restrict__ Qg,   // [B,H,W,C] bf16
    const unsigned short* __restrict__ Kg,
    const unsigned short* __restrict__ Vlg,
    const unsigned short* __restrict__ Vrg,
    const float* __restrict__ lp3_w,         // [C,C]
    const float* __restrict__ lp3_b,
    const float* __restrict__ rp3_w,
    const float* __restrict__ rp3_b,
    unsigned short* __restrict__ P1)         // [B,C,H,W] bf16
{
    extern __shared__ char smem[];
    unsigned short* sQ   = (unsigned short*)(smem);
    unsigned short* sK   = (unsigned short*)(smem + SK_LDS_OFF);
    unsigned short* sVrT = (unsigned short*)(smem + 81920);
    unsigned short* sVlT = (unsigned short*)(smem + 131072);
    float*          sAttn= (float*)(smem + 180224);
    unsigned short* sAw  = (unsigned short*)(smem + 180224);
    unsigned short* sPr  = (unsigned short*)(smem + 205824);
    unsigned short* sPc  = (unsigned short*)(smem + 221184);

    const int t    = threadIdx.x;
    const int wv   = t >> 5;
    const int lane = t & 31;
    const int ml   = (lane >> 4) * 8;
    const int nl   = lane & 15;
    const int bh   = blockIdx.x;
    const int b    = bh / Hh, h = bh % Hh;
    const size_t rowbase = (size_t)bh * SEQ * Cc;

    // ---- stage Q, K: async global->LDS bulk copy (ASYNCcnt), 16B per lane ----
    {
        const char* gq = (const char*)(Qg + rowbase);
        const char* gk = (const char*)(Kg + rowbase);
#pragma unroll
        for (int i = 0; i < 10; ++i) {           // 2560 x 16B per buffer
            unsigned e16 = (unsigned)(t + i * 256) * 16u;
            async_copy16(e16, gq + e16);                 // -> sQ
            async_copy16(SK_LDS_OFF + e16, gk + e16);    // -> sK
        }
    }
    // ---- stage V transposed [c][w] with K-pad to 96 (needs transpose: VALU) ----
    {
        const unsigned short* gvr = Vrg + rowbase;
        const unsigned short* gvl = Vlg + rowbase;
#pragma unroll
        for (int i = 0; i < 80; ++i) {
            int e = t + i * 256;             // 0..20479
            int w = e >> 8, c = e & 255;
            sVrT[c * KP + w] = gvr[e];
            sVlT[c * KP + w] = gvl[e];
        }
#pragma unroll
        for (int i = 0; i < 16; ++i) {       // zero pad w = 80..95
            int e = t + i * 256;             // 0..4095
            int c = e >> 4, w = SEQ + (e & 15);
            sVrT[c * KP + w] = 0;
            sVlT[c * KP + w] = 0;
        }
    }
    wait_async0();                            // ASYNCcnt == 0: sQ/sK resident
    __syncthreads();

    // ---- attn = Ql Kr^T * SCALE ----
    for (int tile = wv; tile < 25; tile += 8) {
        int tm = tile / 5, tn = tile % 5;
        v8f acc = vzero();
#pragma unroll
        for (int kb = 0; kb < Cc; kb += 32) {
            v16bf af = load_frag_a(sQ + tm * 16 * Cc + kb, Cc);
            v16bf bf = load_frag_b(sK + tn * 16 * Cc + kb, Cc);
            acc = wmma_bf16(af, bf, acc);
        }
#pragma unroll
        for (int r = 0; r < 8; ++r)
            sAttn[(tm * 16 + ml + r) * SEQ + tn * 16 + nl] = acc[r] * 0.0625f;
    }
    __syncthreads();

    // ---- dual softmax: t<80 -> row t; 80<=t<160 -> column t-80 (transposed) ----
    if (t < 160) {
        const bool isRow = t < 80;
        const int r = isRow ? t : (t - 80);
        float mx = -3.0e38f;
        for (int j = 0; j < SEQ; ++j) {
            float v = isRow ? sAttn[r * SEQ + j] : sAttn[j * SEQ + r];
            mx = fmaxf(mx, v);
        }
        float sum = 0.f;
        for (int j = 0; j < SEQ; ++j) {
            float v = isRow ? sAttn[r * SEQ + j] : sAttn[j * SEQ + r];
            sum += __expf(v - mx);
        }
        const float inv = 1.f / sum;
        unsigned short* dst = isRow ? (sPr + r * KP) : (sPc + r * KP);
        for (int j = 0; j < SEQ; ++j) {
            float v = isRow ? sAttn[r * SEQ + j] : sAttn[j * SEQ + r];
            dst[j] = f2bf(__expf(v - mx) * inv);
        }
        for (int j = SEQ; j < KP; ++j) dst[j] = 0;
    }
    __syncthreads();

    // ---- Fr2l = Pr @ Vr -> sQ ; Fl2r = Pc @ Vl -> sK  (K = 96 padded) ----
    for (int tile = wv; tile < 80; tile += 8) {
        int tm = tile / 16, tn = tile % 16;      // tm: w-rows, tn: c-cols
        v8f acc = vzero();
        v8f acc2 = vzero();
#pragma unroll
        for (int kb = 0; kb < KP; kb += 32) {
            v16bf ar = load_frag_a(sPr + tm * 16 * KP + kb, KP);
            v16bf br = load_frag_b(sVrT + tn * 16 * KP + kb, KP);
            acc = wmma_bf16(ar, br, acc);
            v16bf al = load_frag_a(sPc + tm * 16 * KP + kb, KP);
            v16bf bl = load_frag_b(sVlT + tn * 16 * KP + kb, KP);
            acc2 = wmma_bf16(al, bl, acc2);
        }
#pragma unroll
        for (int r = 0; r < 8; ++r) {
            sQ[(tm * 16 + ml + r) * Cc + tn * 16 + nl] = f2bf(acc[r]);
            sK[(tm * 16 + ml + r) * Cc + tn * 16 + nl] = f2bf(acc2[r]);
        }
    }
    __syncthreads();

    // ---- fused output projections: pattn1 = lp3@Fr2l^T + rp3@Fl2r^T + b ----
    // M=256 (16 m-tiles), N=80 (5 n-tiles) => 80 tiles, 10 per wave.
    v8f acc[10];
#pragma unroll
    for (int i = 0; i < 10; ++i) acc[i] = vzero();

    for (int pass = 0; pass < 2; ++pass) {
        const float* Wm = pass ? rp3_w : lp3_w;
        const unsigned short* Bsrc = pass ? sK : sQ;
        for (int kb = 0; kb < Cc; kb += 32) {
            __syncthreads();                      // previous sAw consumers done
#pragma unroll
            for (int i = 0; i < 32; ++i) {        // stage As[256][32]
                int e = t + i * 256;              // 0..8191
                int m = e >> 5, k = e & 31;
                sAw[e] = f2bf(Wm[m * Cc + kb + k]);
            }
            __syncthreads();
            int idx = 0;
            for (int tile = wv; tile < 80; tile += 8, ++idx) {
                int tm = tile / 5, tn = tile % 5;
                v16bf af = load_frag_a(sAw + tm * 16 * 32, 32);
                v16bf bf = load_frag_b(Bsrc + tn * 16 * Cc + kb, Cc);
                acc[idx] = wmma_bf16(af, bf, acc[idx]);
            }
        }
    }

    {
        int idx = 0;
        for (int tile = wv; tile < 80; tile += 8, ++idx) {
            int tm = tile / 5, tn = tile % 5;
#pragma unroll
            for (int r = 0; r < 8; ++r) {
                int co = tm * 16 + ml + r;
                int w  = tn * 16 + nl;
                float v = acc[idx][r] + lp3_b[co] + rp3_b[co];
                P1[((size_t)(b * Cc + co) * Hh + h) * Ww + w] = f2bf(v);
            }
        }
    }
}

// ---------------------------------------------------------------------------
// Kernel 4: PixelAttention 7x7 grouped conv (reflect pad) + double sigmoid
// + final blend.  One block = one channel plane's 16x16 tile, halo in LDS.
// ---------------------------------------------------------------------------
__device__ __forceinline__ int refl(int i, int n) {
    if (i < 0) i = -i;
    if (i >= n) i = 2 * n - 2 - i;
    return i;
}

__global__ __launch_bounds__(256) void pixattn_kernel(
    const float* __restrict__ xl,
    const float* __restrict__ xr,
    const unsigned short* __restrict__ P1,   // [B,C,H,W] bf16
    const float* __restrict__ paw,           // [C,2,7,7]
    const float* __restrict__ pab,           // [C]
    float* __restrict__ out)                 // [B,C,H,W] f32
{
    __shared__ float t0[22 * 22];
    __shared__ float t1[22 * 22];
    __shared__ float wloc[98];

    const int t  = threadIdx.x;
    const int bc = blockIdx.z;               // b*256 + c
    const int c  = bc & 255;
    const int h0 = blockIdx.y * 16;
    const int w0 = blockIdx.x * 16;

    if (t < 98) wloc[t] = paw[c * 98 + t];

    const size_t pbase = (size_t)bc * HWp;
    for (int e = t; e < 22 * 22; e += 256) {
        int ly = e / 22, lx = e % 22;
        int gy = refl(h0 + ly - 3, Hh);
        int gx = refl(w0 + lx - 3, Ww);
        size_t g = pbase + gy * Ww + gx;
        t0[e] = xl[g] + xr[g];               // "initial"
        t1[e] = bf2f(P1[g]);                 // pattn1
    }
    __syncthreads();

    const int ty = t >> 4, tx = t & 15;
    float s = pab[c];
#pragma unroll
    for (int dy = 0; dy < 7; ++dy)
#pragma unroll
        for (int dx = 0; dx < 7; ++dx)
            s += wloc[dy * 7 + dx]      * t0[(ty + dy) * 22 + tx + dx]
               + wloc[49 + dy * 7 + dx] * t1[(ty + dy) * 22 + tx + dx];

    float a = 1.f / (1.f + __expf(-s));
    a = 1.f / (1.f + __expf(-a));
    const size_t g = pbase + (size_t)(h0 + ty) * Ww + (w0 + tx);
    out[g] = a * xl[g] + (1.f - a) * xr[g];
}

// ---------------------------------------------------------------------------
// Host launch.  Workspace layout (bf16 elems, SZ = 16*80*80*256 = 26,214,400):
//   [0,SZ)    Q    [B,H,W,C]
//   [SZ,2SZ)  K    [B,H,W,C]
//   [2SZ,3SZ) Vl   [B,H,W,C]
//   [3SZ,4SZ) Vr   [B,H,W,C]
//   [4SZ,5SZ) pattn1 [B,C,H,W]
//   [5SZ,6SZ) tmp  [B,C,H,W]  (conv1x1 out, reused across the 4 projections)
// Total 6*SZ*2 = 314,572,800 bytes.
// ---------------------------------------------------------------------------
extern "C" void kernel_launch(void* const* d_in, const int* in_sizes, int n_in,
                              void* d_out, int out_size, void* d_ws, size_t ws_size,
                              hipStream_t stream) {
    const float* x_l    = (const float*)d_in[0];
    const float* x_r    = (const float*)d_in[1];
    const float* lp1_w1 = (const float*)d_in[2];
    const float* lp1_b1 = (const float*)d_in[3];
    const float* lp1_w2 = (const float*)d_in[4];
    const float* lp1_b2 = (const float*)d_in[5];
    const float* rp1_w1 = (const float*)d_in[6];
    const float* rp1_b1 = (const float*)d_in[7];
    const float* rp1_w2 = (const float*)d_in[8];
    const float* rp1_b2 = (const float*)d_in[9];
    const float* lp2_w1 = (const float*)d_in[10];
    const float* lp2_b1 = (const float*)d_in[11];
    const float* lp2_w2 = (const float*)d_in[12];
    const float* lp2_b2 = (const float*)d_in[13];
    const float* rp2_w1 = (const float*)d_in[14];
    const float* rp2_b1 = (const float*)d_in[15];
    const float* rp2_w2 = (const float*)d_in[16];
    const float* rp2_b2 = (const float*)d_in[17];
    const float* lp3_w  = (const float*)d_in[18];
    const float* lp3_b  = (const float*)d_in[19];
    const float* rp3_w  = (const float*)d_in[20];
    const float* rp3_b  = (const float*)d_in[21];
    const float* pa_w   = (const float*)d_in[22];
    const float* pa_b   = (const float*)d_in[23];
    float* outp = (float*)d_out;

    const size_t SZ = (size_t)Bn * Hh * Ww * Cc;
    unsigned short* ws   = (unsigned short*)d_ws;
    unsigned short* Qb   = ws;
    unsigned short* Kb   = ws + SZ;
    unsigned short* Vlb  = ws + 2 * SZ;
    unsigned short* Vrb  = ws + 3 * SZ;
    unsigned short* P1b  = ws + 4 * SZ;
    unsigned short* tmpb = ws + 5 * SZ;

    const dim3 blk(256);
    const dim3 gGemm(HWp / 64, Cc / 128, Bn);        // 100 x 2 x 16
    const dim3 gDw(Ww / 16, Hh, Bn * (Cc / 16));     // 5 x 80 x 256

    // Four projections (stream-ordered; tmpb reused)
    conv1x1_kernel<<<gGemm, blk, 0, stream>>>(x_l, lp1_w1, lp1_b1, tmpb);
    dw3x3_kernel<<<gDw, blk, 0, stream>>>(tmpb, lp1_w2, lp1_b2, Qb);
    conv1x1_kernel<<<gGemm, blk, 0, stream>>>(x_r, rp1_w1, rp1_b1, tmpb);
    dw3x3_kernel<<<gDw, blk, 0, stream>>>(tmpb, rp1_w2, rp1_b2, Kb);
    conv1x1_kernel<<<gGemm, blk, 0, stream>>>(x_l, lp2_w1, lp2_b1, tmpb);
    dw3x3_kernel<<<gDw, blk, 0, stream>>>(tmpb, lp2_w2, lp2_b2, Vlb);
    conv1x1_kernel<<<gGemm, blk, 0, stream>>>(x_r, rp2_w1, rp2_b1, tmpb);
    dw3x3_kernel<<<gDw, blk, 0, stream>>>(tmpb, rp2_w2, rp2_b2, Vrb);

    // Fused attention + output projections (one WG per (b,h))
    attn_kernel<<<Bn * Hh, blk, ATTN_LDS_BYTES, stream>>>(
        Qb, Kb, Vlb, Vrb, lp3_w, lp3_b, rp3_w, rp3_b, P1b);

    // PixelAttention + blend
    pixattn_kernel<<<dim3(Ww / 16, Hh / 16, Bn * Cc), blk, 0, stream>>>(
        x_l, x_r, P1b, pa_w, pa_b, outp);
}
